// MHA_57724360458398
// MI455X (gfx1250) — matrix operand order repaired and checked by reading, not compiled, not run on hardware
//
#include <hip/hip_runtime.h>
#include <hip/hip_bf16.h>
#include <math.h>

// ---------------------------------------------------------------------------
// MHA forward for MI455X (gfx1250, wave32, WMMA).
// B=4, N=2048, C=1024, H=16, D=64.
//   - all GEMMs on v_wmma_f32_16x16x32_bf16 (fp32 accumulate)
//   - weights pre-transposed+converted to bf16 so every WMMA operand lane
//     reads 16 contiguous bf16 (2x b128)
//   - flash attention: per-block cooperative K/V tile staging in LDS via
//     global_load_async_to_lds_b128 (ASYNCcnt), double-buffered
// ---------------------------------------------------------------------------

typedef __attribute__((ext_vector_type(16))) __bf16 v16bf;
typedef __attribute__((ext_vector_type(2)))  __bf16 bf16x2;
typedef __attribute__((ext_vector_type(8)))  float  v8f;

#define WMMA_BF16(a, b, c) \
  __builtin_amdgcn_wmma_f32_16x16x32_bf16(false, (a), false, (b), (short)0, (c), false, false)

// A operand (16x32, 16-bit): lane holds row lane&15; element e has
// K = e + 16*(lane>>4)  -> 16 contiguous elements per lane (matches B).
__device__ __forceinline__ v16bf load_afrag(const __bf16* p, int half) {
  v16bf f;
#pragma unroll
  for (int e = 0; e < 16; e += 2) {
    const int k = e + 16 * half;
    const bf16x2 v = *reinterpret_cast<const bf16x2*>(p + k);
    f[e] = v[0];
    f[e + 1] = v[1];
  }
  return f;
}

// B operand (32x16, 16-bit), contiguous k order (same as A): lane holds column lane&15;
// element e has K = e + 16*(lane>>4) -> 16 contiguous elements per lane.
__device__ __forceinline__ v16bf load_bfrag(const __bf16* p, int half) {
  v16bf f;
  const __bf16* q = p + 16 * half;
#pragma unroll
  for (int e = 0; e < 16; e += 2) {
    const bf16x2 v = *reinterpret_cast<const bf16x2*>(q + e);
    f[e] = v[0];
    f[e + 1] = v[1];
  }
  return f;
}

// ---------------------------------------------------------------------------
// Prep: fp32 -> bf16 convert (x), and 4x fused transpose+convert (weights).
// ---------------------------------------------------------------------------
__global__ __launch_bounds__(256) void convert_bf16_kernel(
    const float* __restrict__ X, __bf16* __restrict__ Y, int n) {
  const int i = (blockIdx.x * 256 + threadIdx.x) * 4;
  if (i < n) {
    const float4 v = *reinterpret_cast<const float4*>(X + i);
    Y[i] = (__bf16)v.x;
    Y[i + 1] = (__bf16)v.y;
    Y[i + 2] = (__bf16)v.z;
    Y[i + 3] = (__bf16)v.w;
  }
}

__global__ __launch_bounds__(256) void transpose_convert_kernel(
    const float* __restrict__ W0, const float* __restrict__ W1,
    const float* __restrict__ W2, const float* __restrict__ W3,
    __bf16* __restrict__ T0, __bf16* __restrict__ T1,
    __bf16* __restrict__ T2, __bf16* __restrict__ T3, int NK) {
  const float* W = blockIdx.z == 0 ? W0 : blockIdx.z == 1 ? W1
                 : blockIdx.z == 2 ? W2 : W3;
  __bf16* T = blockIdx.z == 0 ? T0 : blockIdx.z == 1 ? T1
            : blockIdx.z == 2 ? T2 : T3;
  __shared__ float tile[32][33];
  const int bx = blockIdx.x * 32, by = blockIdx.y * 32;
#pragma unroll
  for (int j = threadIdx.y; j < 32; j += 8)
    tile[j][threadIdx.x] = W[(size_t)(by + j) * NK + bx + threadIdx.x];
  __syncthreads();
#pragma unroll
  for (int j = threadIdx.y; j < 32; j += 8)
    T[(size_t)(bx + j) * NK + by + threadIdx.x] = (__bf16)tile[threadIdx.x][j];
}

// ---------------------------------------------------------------------------
// GEMM + bias: Y = (X[M,K] @ Wt[N,K]^T + bias) * postScale.
// One wave = 16(M) x 64(N) tile; 8 waves / 256-thread block.
// MODE 0: fp32 [M,N]; MODE 1: bf16 [M,N]; MODE 2: bf16 per-head transposed
//         Vt[((b*16+h)*64+d)*2048 + seq]  (for the attention P@V operand).
// ---------------------------------------------------------------------------
template <int MODE>
__global__ __launch_bounds__(256) void gemm_bt_kernel(
    const __bf16* __restrict__ X, const __bf16* __restrict__ Wt,
    const float* __restrict__ bias, void* __restrict__ Yv,
    int M, int N, int K, float postScale) {
  const int lane = threadIdx.x & 31;
  const int wave = threadIdx.x >> 5;
  const int half = lane >> 4;
  const int r16  = lane & 15;

  const int tilesN = N >> 6;
  const int gw = blockIdx.x * 8 + wave;
  const int m0 = (gw / tilesN) * 16;
  const int n0 = (gw % tilesN) * 64;

  v8f acc[4] = {};

  for (int k0 = 0; k0 < K; k0 += 32) {
    const v16bf a = load_afrag(X + (size_t)(m0 + r16) * K + k0, half);
#pragma unroll
    for (int c = 0; c < 4; ++c) {
      const v16bf b =
          load_bfrag(Wt + (size_t)(n0 + c * 16 + r16) * K + k0, half);
      acc[c] = WMMA_BF16(a, b, acc[c]);
    }
  }

#pragma unroll
  for (int c = 0; c < 4; ++c) {
    const int n = n0 + c * 16 + r16;
    const float bv = bias[n];
#pragma unroll
    for (int r = 0; r < 8; ++r) {
      const int m = m0 + r + 8 * half;
      const float y = (acc[c][r] + bv) * postScale;
      if (MODE == 0) {
        ((float*)Yv)[(size_t)m * N + n] = y;
      } else if (MODE == 1) {
        ((__bf16*)Yv)[(size_t)m * N + n] = (__bf16)y;
      } else {
        const int bi = m >> 11, seq = m & 2047;  // token -> (batch, position)
        ((__bf16*)Yv)[(((size_t)bi * 16 + (n >> 6)) * 64 + (n & 63)) * 2048 +
                      seq] = (__bf16)y;
      }
    }
  }
}

// ---------------------------------------------------------------------------
// Flash attention. Block = 4 waves sharing one (b,h); each wave owns 16 query
// rows. K (32x64) and V^T (64x32) tiles are async-staged into LDS once per
// block (double-buffered), cutting K/V traffic 4x. Online softmax via 16-lane
// shuffle reductions; P re-layout (C->A) through a per-wave LDS tile.
// Q is pre-scaled by log2(e)/sqrt(D) so softmax uses exp2.
// ---------------------------------------------------------------------------
__global__ __launch_bounds__(128) void flash_attn_kernel(
    const __bf16* __restrict__ Q, const __bf16* __restrict__ Km,
    const __bf16* __restrict__ Vt, __bf16* __restrict__ O) {
  const int lane = threadIdx.x & 31;
  const int wave = threadIdx.x >> 5;
  const int half = lane >> 4;
  const int r16  = lane & 15;

  // Padded rows (72/40/40 elems) to spread LDS banks; 16B-aligned for b128.
  __shared__ __align__(16) __bf16 kbuf[2][32 * 72];  // K tile: 32 keys x 64 d
  __shared__ __align__(16) __bf16 vbuf[2][64 * 40];  // V^T tile: 64 d x 32 keys
  __shared__ __align__(16) __bf16 pbuf[4][16 * 40];  // per-wave P re-layout
  __bf16* myP = pbuf[wave];

  const int gw = blockIdx.x * 4 + wave;
  const int m0 = (gw & 127) * 16;  // query tile in sequence
  const int bh = gw >> 7;          // identical for all 4 waves of the block
  const int b = bh >> 4, h = bh & 15;

  const size_t tokBase = ((size_t)b * 2048) * 1024 + (size_t)h * 64;
  const __bf16* Qp = Q + tokBase;
  const __bf16* Kp = Km + tokBase;
  const __bf16* Vp = Vt + (size_t)bh * 64 * 2048;

  // Q fragments (16 x 64 = two 32-wide K-steps), loaded once.
  v16bf qf[2];
#pragma unroll
  for (int s = 0; s < 2; ++s)
    qf[s] = load_afrag(Qp + (size_t)(m0 + r16) * 1024 + s * 32, half);

  float mrow[8], lrow[8];
  v8f o[4] = {};
#pragma unroll
  for (int r = 0; r < 8; ++r) {
    mrow[r] = -__builtin_huge_valf();
    lrow[r] = 0.0f;
  }

  // Cooperative async staging of one K tile (32x128B) + one V^T tile (64x64B).
  auto issue_tile = [&](int kb, int bufi) {
    const unsigned kl = (unsigned)(size_t)&kbuf[bufi][0];
    const unsigned vl = (unsigned)(size_t)&vbuf[bufi][0];
#pragma unroll
    for (int i = 0; i < 2; ++i) {  // K: 256 16B chunks over 128 threads
      const int chunk = threadIdx.x + i * 128;
      const int row = chunk >> 3, off = chunk & 7;
      const __bf16* gp = Kp + (size_t)(kb * 32 + row) * 1024 + off * 8;
      const unsigned l = kl + row * 144 + off * 16;
      asm volatile("global_load_async_to_lds_b128 %0, %1, off" ::"v"(l),
                   "v"((unsigned long long)(size_t)gp)
                   : "memory");
    }
#pragma unroll
    for (int i = 0; i < 2; ++i) {  // V^T: 256 16B chunks over 128 threads
      const int chunk = threadIdx.x + i * 128;
      const int row = chunk >> 2, off = chunk & 3;
      const __bf16* gp = Vp + (size_t)row * 2048 + kb * 32 + off * 8;
      const unsigned l = vl + row * 80 + off * 16;
      asm volatile("global_load_async_to_lds_b128 %0, %1, off" ::"v"(l),
                   "v"((unsigned long long)(size_t)gp)
                   : "memory");
    }
  };

  issue_tile(0, 0);

  for (int kb = 0; kb < 64; ++kb) {
    const int cur = kb & 1;
    if (kb + 1 < 64) {
      issue_tile(kb + 1, cur ^ 1);
      // In-order async completion: <=4 outstanding == current tile landed.
      asm volatile("s_wait_asynccnt 0x4" ::: "memory");
    } else {
      asm volatile("s_wait_asynccnt 0x0" ::: "memory");
    }
    __syncthreads();  // all waves' chunks of the current tile visible

    const __bf16* kt = kbuf[cur];
    const __bf16* vt = vbuf[cur];

    // S = Q @ K^T : 16 queries x 32 keys (two 16-wide key chunks).
    v8f sfr[2] = {};
#pragma unroll
    for (int c = 0; c < 2; ++c)
#pragma unroll
      for (int s = 0; s < 2; ++s) {
        const v16bf kf = load_bfrag(kt + (c * 16 + r16) * 72 + s * 32, half);
        sfr[c] = WMMA_BF16(qf[s], kf, sfr[c]);
      }

    // Online softmax; row m = r + 8*half spans the 16 lanes of a half-wave.
    float corr[8];
#pragma unroll
    for (int r = 0; r < 8; ++r) {
      float v = fmaxf(sfr[0][r], sfr[1][r]);
      v = fmaxf(v, __shfl_xor(v, 8, 16));
      v = fmaxf(v, __shfl_xor(v, 4, 16));
      v = fmaxf(v, __shfl_xor(v, 2, 16));
      v = fmaxf(v, __shfl_xor(v, 1, 16));
      const float mn = fmaxf(mrow[r], v);
      corr[r] = exp2f(mrow[r] - mn);
      mrow[r] = mn;
      const float p0 = exp2f(sfr[0][r] - mn);
      const float p1 = exp2f(sfr[1][r] - mn);
      myP[(r + 8 * half) * 40 + r16] = (__bf16)p0;
      myP[(r + 8 * half) * 40 + 16 + r16] = (__bf16)p1;
      float ps = p0 + p1;
      ps += __shfl_xor(ps, 8, 16);
      ps += __shfl_xor(ps, 4, 16);
      ps += __shfl_xor(ps, 2, 16);
      ps += __shfl_xor(ps, 1, 16);
      lrow[r] = lrow[r] * corr[r] + ps;
    }
#pragma unroll
    for (int dc = 0; dc < 4; ++dc)
#pragma unroll
      for (int r = 0; r < 8; ++r) o[dc][r] *= corr[r];

    // Per-wave LDS ordering: P stores before A-layout reload.
    asm volatile("s_wait_dscnt 0x0" ::: "memory");
    const v16bf pf = load_afrag(myP + r16 * 40, half);

    // O += P @ V : four 16-wide d chunks; V^T rows are key-contiguous.
#pragma unroll
    for (int dc = 0; dc < 4; ++dc) {
      const v16bf vf = load_bfrag(vt + (dc * 16 + r16) * 40, half);
      o[dc] = WMMA_BF16(pf, vf, o[dc]);
    }
    __syncthreads();  // tile fully consumed before it is overwritten
  }

  // Normalize, store bf16 [B*N, C] (feeds the bf16 output projection).
#pragma unroll
  for (int r = 0; r < 8; ++r) lrow[r] = 1.0f / lrow[r];
#pragma unroll
  for (int dc = 0; dc < 4; ++dc)
#pragma unroll
    for (int r = 0; r < 8; ++r) {
      const size_t row = (size_t)b * 2048 + m0 + r + 8 * half;
      O[row * 1024 + h * 64 + dc * 16 + r16] = (__bf16)(o[dc][r] * lrow[r]);
    }
}

// ---------------------------------------------------------------------------
extern "C" void kernel_launch(void* const* d_in, const int* in_sizes, int n_in,
                              void* d_out, int out_size, void* d_ws,
                              size_t ws_size, hipStream_t stream) {
  const float* x  = (const float*)d_in[0];
  const float* Wq = (const float*)d_in[1];
  const float* bq = (const float*)d_in[2];
  const float* Wk = (const float*)d_in[3];
  const float* bk = (const float*)d_in[4];
  const float* Wv = (const float*)d_in[5];
  const float* bv = (const float*)d_in[6];
  const float* Wo = (const float*)d_in[7];
  const float* bo = (const float*)d_in[8];
  float* out = (float*)d_out;

  const int M = 4 * 2048;  // B*N tokens
  const int C = 1024;
  const size_t MB = (size_t)1 << 20;

  // Workspace: xb 16MB | 4x Wt bf16 2MB | Q 16MB | K 16MB | Vt 16MB | AO 16MB
  char* ws = (char*)d_ws;
  __bf16* xb  = (__bf16*)(ws);
  __bf16* Wqt = (__bf16*)(ws + 16 * MB);
  __bf16* Wkt = (__bf16*)(ws + 18 * MB);
  __bf16* Wvt = (__bf16*)(ws + 20 * MB);
  __bf16* Wot = (__bf16*)(ws + 22 * MB);
  __bf16* Qw  = (__bf16*)(ws + 24 * MB);
  __bf16* Kw  = (__bf16*)(ws + 40 * MB);
  __bf16* Vtw = (__bf16*)(ws + 56 * MB);
  __bf16* AO  = (__bf16*)(ws + 72 * MB);

  // Prep: convert x, transpose+convert all 4 weights.
  convert_bf16_kernel<<<dim3((M * C) / 4 / 256), dim3(256), 0, stream>>>(
      x, xb, M * C);
  transpose_convert_kernel<<<dim3(32, 32, 4), dim3(32, 8), 0, stream>>>(
      Wq, Wk, Wv, Wo, Wqt, Wkt, Wvt, Wot, C);

  const dim3 gblk(256);
  const dim3 ggrd((unsigned)((M / 16) * (C / 64) / 8));  // 1024 blocks

  // Fold 1/sqrt(D) and log2(e) (exp2-based softmax) into Q.
  const float qscale = 0.125f * 1.4426950408889634f;

  gemm_bt_kernel<1><<<ggrd, gblk, 0, stream>>>(xb, Wqt, bq, Qw, M, C, C, qscale);
  gemm_bt_kernel<1><<<ggrd, gblk, 0, stream>>>(xb, Wkt, bk, Kw, M, C, C, 1.0f);
  gemm_bt_kernel<2><<<ggrd, gblk, 0, stream>>>(xb, Wvt, bv, Vtw, M, C, C, 1.0f);

  // waves = B*H*(N/16) = 8192 -> 2048 blocks of 4 waves (one (b,h) per block).
  flash_attn_kernel<<<dim3(2048), dim3(128), 0, stream>>>(Qw, Kw, Vtw, AO);

  gemm_bt_kernel<0><<<ggrd, gblk, 0, stream>>>(AO, Wot, bo, out, M, C, C, 1.0f);
}
